// InfiniAttention_64725157151131
// MI455X (gfx1250) — matrix operand order repaired and checked by reading, not compiled, not run on hardware
//
#include <hip/hip_runtime.h>

// ---------------- problem constants ----------------
#define B_   2
#define S_   2048
#define HID_ 2048
#define H_   16
#define D_   128
#define BS_  (B_ * S_)          // 4096
#define BHSD (BS_ * HID_)       // 8,388,608 (== B*H*S*D since H*D == HID)

// ---------------- CDNA5 WMMA types ----------------
typedef __attribute__((ext_vector_type(16))) __bf16 v16bf;
typedef __attribute__((ext_vector_type(8)))  float  v8f;

union Frag {
    v16bf v;
    uint4 q[2];
    unsigned short us[16];
};

__device__ __forceinline__ unsigned short f2bf(float f) {
    unsigned int u; __builtin_memcpy(&u, &f, 4);
    u += 0x7FFFu + ((u >> 16) & 1u);      // round-to-nearest-even
    return (unsigned short)(u >> 16);
}
__device__ __forceinline__ float bf2f(unsigned short h) {
    unsigned int u = ((unsigned int)h) << 16;
    float f; __builtin_memcpy(&f, &u, 4); return f;
}
__device__ __forceinline__ v8f wmma_bf16(const Frag& a, const Frag& b, v8f c) {
    // D = A(16x32 bf16) * B(32x16 bf16) + C(16x16 f32)
    return __builtin_amdgcn_wmma_f32_16x16x32_bf16(
        false, a.v, false, b.v, (short)0, c, false, false);
}
__device__ __forceinline__ v8f vzero8() {
    v8f z = {0.f,0.f,0.f,0.f,0.f,0.f,0.f,0.f};
    return z;
}

// ---- CDNA5 async Global->LDS copy (ASYNCcnt-tracked), 32 bytes per lane ----
// INST_OFFSET is added to BOTH the LDS and global addresses, so one address
// pair serves both 16B halves.
__device__ __forceinline__ void async_ld32B(unsigned lds_off, const void* g) {
    unsigned long long ga = (unsigned long long)g;
    asm volatile("global_load_async_to_lds_b128 %0, %1, off"
                 :: "v"(lds_off), "v"(ga) : "memory");
    asm volatile("global_load_async_to_lds_b128 %0, %1, off offset:16"
                 :: "v"(lds_off), "v"(ga) : "memory");
}
__device__ __forceinline__ void wait_async0() {
    asm volatile("s_wait_asynccnt 0x0" ::: "memory");
}
// async loads complete in order: waiting <=4 with 8 outstanding guarantees the
// oldest tile's 4 chunks have landed while the next tile stays in flight.
__device__ __forceinline__ void wait_async4() {
    asm volatile("s_wait_asynccnt 0x4" ::: "memory");
}
__device__ __forceinline__ unsigned lds_addr_of(const void* p) {
    // generic pointer to LDS: low 32 bits hold the LDS byte offset
    return (unsigned)(unsigned long long)p;
}

// ---------------- conversion kernels ----------------
__global__ __launch_bounds__(256) void cvt_bf16_kernel(
    const float* __restrict__ src, unsigned short* __restrict__ dst, int n) {
    int i = blockIdx.x * 256 + threadIdx.x;
    if (i < n) dst[i] = f2bf(src[i]);
}

// out[batch][c][r] = in[batch][r][c]  (bf16)
__global__ __launch_bounds__(256) void cvt_bf16_T_kernel(
    const float* __restrict__ src, unsigned short* __restrict__ dst,
    int rows, int cols) {
    int n = rows * cols;
    int i = blockIdx.x * 256 + threadIdx.x;
    if (i >= n) return;
    int r = i / cols, c = i % cols;
    size_t base = (size_t)blockIdx.y * (size_t)n;
    dst[base + (size_t)c * rows + r] = f2bf(src[base + i]);
}

// ---------------- WMMA GEMM: C[M,N] = A[M,K](bf16) * Wt[N,K](bf16) + bias ----
// block = 256 threads (8 waves), tile 128x128, K-step 32.
// 3-buffer async Global->LDS pipeline (2 tiles in flight), one barrier per
// K-step. All 9 fragments are loaded before the 8 WMMAs so LDS latency is paid
// once per tile and the WMMAs stream back-to-back into the XDL pipe.
__global__ __launch_bounds__(256) void gemm_bf16_kernel(
    const unsigned short* __restrict__ A,
    const unsigned short* __restrict__ Wt,
    const float* __restrict__ bias,
    float* __restrict__ C, int M, int N, int K) {
    __shared__ unsigned short sA[3][128 * 32];  // [buf][row][k]
    __shared__ unsigned short sB[3][128 * 32];  // [buf][col][k]
    const int tid  = threadIdx.x;
    const int lane = tid & 31, wave = tid >> 5;
    const int lr = lane & 15, lh = lane >> 4;
    const int gm0 = blockIdx.y * 128, gn0 = blockIdx.x * 128;
    const int row = tid >> 1, half = tid & 1;

    const unsigned short* gA = A  + (size_t)(gm0 + row) * K + half * 16;
    const unsigned short* gB = Wt + (size_t)(gn0 + row) * K + half * 16;
    unsigned ldsA[3], ldsB[3];
    for (int bfi = 0; bfi < 3; bfi++) {
        ldsA[bfi] = lds_addr_of(&sA[bfi][row * 32 + half * 16]);
        ldsB[bfi] = lds_addr_of(&sB[bfi][row * 32 + half * 16]);
    }

    v8f acc[8];
    for (int i = 0; i < 8; i++) acc[i] = vzero8();

    const int aoff = (wave * 16 + lr) * 32 + lh * 16; // A-frag base in tile
    const int boff = lr * 32 + lh * 16;               // B-frag base in tile

    // fragment-batched compute on buffer cb: 18 ds_loads, then 8 WMMAs
    auto compute = [&](int cb) {
        Frag fa, fb[8];
        const unsigned short* pa = &sA[cb][aoff];
        fa.q[0] = *(const uint4*)(pa);
        fa.q[1] = *(const uint4*)(pa + 8);
        for (int nt = 0; nt < 8; nt++) {
            const unsigned short* pb = &sB[cb][boff + nt * (16 * 32)];
            fb[nt].q[0] = ((const uint4*)pb)[0];
            fb[nt].q[1] = ((const uint4*)pb)[1];
        }
        for (int nt = 0; nt < 8; nt++)
            acc[nt] = wmma_bf16(fa, fb[nt], acc[nt]);
    };

    const int T = K / 32;  // K is a multiple of 32 and >= 96 for all callers
    // prologue: stage tiles 0 and 1
    async_ld32B(ldsA[0], gA);
    async_ld32B(ldsB[0], gB);
    async_ld32B(ldsA[1], gA + 32);
    async_ld32B(ldsB[1], gB + 32);

    int bufC = 0;  // rotating compute buffer (no integer modulo)
    for (int i = 0; i < T - 1; i++) {
        wait_async4();      // tile i landed; tile i+1 still in flight
        __syncthreads();    // all waves landed; compute(i-1) done (WAR-safe)
        if (i + 2 < T) {    // keep two tiles in flight
            int bufN = bufC + 2; if (bufN >= 3) bufN -= 3;
            async_ld32B(ldsA[bufN], gA + (size_t)(i + 2) * 32);
            async_ld32B(ldsB[bufN], gB + (size_t)(i + 2) * 32);
        }
        compute(bufC);
        bufC = (bufC == 2) ? 0 : bufC + 1;
    }
    // peeled last iteration: only tile T-1 in flight
    wait_async0();
    __syncthreads();
    compute(bufC);

    for (int nt = 0; nt < 8; nt++) {
        int gn = gn0 + nt * 16 + lr;
        float bv = bias ? bias[gn] : 0.f;
        for (int r = 0; r < 8; r++) {
            int gm = gm0 + wave * 16 + r + (lh ? 8 : 0);
            C[(size_t)gm * N + gn] = acc[nt][r] + bv;
        }
    }
}

// ---------------- RoPE + ELU feature map + layout shuffles ----------------
// one block per (b,h,s), 128 threads = D
__global__ __launch_bounds__(128) void rope_feat_kernel(
    const float* __restrict__ qp, const float* __restrict__ kp,
    const float* __restrict__ vp, const float* __restrict__ z,
    unsigned short* __restrict__ q_r, unsigned short* __restrict__ k_r,
    unsigned short* __restrict__ sig_q, unsigned short* __restrict__ sig_k_t,
    unsigned short* __restrict__ v_t, float* __restrict__ mem_den) {
    int idx = blockIdx.x;
    int s  = idx % S_;
    int hb = idx / S_;
    int h  = hb % H_;
    int b  = hb / H_;
    int d  = threadIdx.x;

    size_t src = (size_t)(b * S_ + s) * HID_ + h * D_ + d;
    float qv = qp[src], kv = kp[src], vv = vp[src];

    __shared__ float shq[D_], shk[D_];
    shq[d] = qv; shk[d] = kv;
    __syncthreads();
    float oq = (d < D_ / 2) ? -shq[d + D_ / 2] : shq[d - D_ / 2];
    float ok = (d < D_ / 2) ? -shk[d + D_ / 2] : shk[d - D_ / 2];

    int i = d & (D_ / 2 - 1);
    // inv_freq = 10000^(-2i/128) = exp(-2i*ln(10000)/128)
    float inv = __expf(-(float)(2 * i) * (9.210340371976184f / (float)D_));
    float ang = (float)s * inv;
    float sn, cs; __sincosf(ang, &sn, &cs);
    float qr = qv * cs + oq * sn;
    float kr = kv * cs + ok * sn;
    float sq = qr > 0.f ? qr + 1.f : __expf(qr);   // elu(x)+1
    float sk = kr > 0.f ? kr + 1.f : __expf(kr);

    size_t bh   = (size_t)(b * H_ + h);
    size_t o_sd = (bh * S_ + s) * D_ + d;   // [B,H,S,D]
    size_t o_ds = (bh * D_ + d) * S_ + s;   // [B,H,D,S]
    q_r[o_sd]     = f2bf(qr);
    k_r[o_sd]     = f2bf(kr);
    sig_q[o_sd]   = f2bf(sq);
    sig_k_t[o_ds] = f2bf(sk);
    v_t[o_ds]     = f2bf(vv);

    __shared__ float red[D_];
    red[d] = sq * z[h * D_ + d];
    __syncthreads();
    for (int st = 64; st > 0; st >>= 1) {
        if (d < st) red[d] += red[d + st];
        __syncthreads();
    }
    if (d == 0) mem_den[bh * S_ + s] = red[0];
}

// ---------------- z_new = z + sum_{b,s} sig_k -------------------------------
__global__ __launch_bounds__(256) void znew_kernel(
    const unsigned short* __restrict__ sig_k_t, const float* __restrict__ z,
    float* __restrict__ z_out) {
    int row = blockIdx.x;  // h*D + d
    float acc = 0.f;
    for (int b = 0; b < B_; b++) {
        const unsigned short* p = sig_k_t + ((size_t)(b * H_ * D_) + row) * S_;
        for (int s = threadIdx.x; s < S_; s += 256) acc += bf2f(p[s]);
    }
    __shared__ float red[256];
    red[threadIdx.x] = acc;
    __syncthreads();
    for (int st = 128; st > 0; st >>= 1) {
        if (threadIdx.x < st) red[threadIdx.x] += red[threadIdx.x + st];
        __syncthreads();
    }
    if (threadIdx.x == 0) z_out[row] = z[row] + red[0];
}

// ---------------- fused flash attention + memory retrieval ------------------
// 8 waves per block, each wave owns one 16-query tile of one (b,h)
__global__ __launch_bounds__(256) void attn_ret_kernel(
    const unsigned short* __restrict__ q_r, const unsigned short* __restrict__ k_r,
    const unsigned short* __restrict__ v_t, const unsigned short* __restrict__ sig_q,
    const unsigned short* __restrict__ M_t, const float* __restrict__ mem_den,
    const float* __restrict__ beta, unsigned short* __restrict__ comb) {
    __shared__ unsigned short sP[8][16 * 32];  // per-wave P staging (C->A relayout)
    const int lane = threadIdx.x & 31, wave = threadIdx.x >> 5;
    const int lr = lane & 15, lh = lane >> 4;
    const int gw = blockIdx.x * 8 + wave;
    const int qt = gw & (S_ / 16 - 1);
    const int bh = gw / (S_ / 16);
    const int h  = bh % H_;
    const int b  = bh / H_;
    const int q0 = qt * 16;
    const float scale = 0.08838834764831845f;  // 1/sqrt(128)

    // Q fragments (A layout): lane holds row q0+lr, all 128 d in 4 frags
    Frag qf[4];
    const unsigned short* qbase = q_r + ((size_t)bh * S_ + q0 + lr) * D_;
    for (int dc = 0; dc < 4; dc++) {
        qf[dc].q[0] = *(const uint4*)(qbase + dc * 32 + lh * 16);
        qf[dc].q[1] = *(const uint4*)(qbase + dc * 32 + 8 + lh * 16);
    }

    v8f O[8];
    float m[8], l[8], scarr[8];
    for (int i = 0; i < 8; i++) { O[i] = vzero8(); m[i] = -1e30f; l[i] = 0.f; }
    unsigned short* myP = sP[wave];

    for (int kb = 0; kb <= q0; kb += 32) {
        // scores: two 16x16 tiles (keys kb..kb+15, kb+16..kb+31)
        v8f s0 = vzero8(), s1 = vzero8();
        for (int dc = 0; dc < 4; dc++) {
            Frag kf0, kf1;
            const unsigned short* p0 = k_r + ((size_t)bh * S_ + kb + lr) * D_ + dc * 32 + lh * 16;
            const unsigned short* p1 = k_r + ((size_t)bh * S_ + kb + 16 + lr) * D_ + dc * 32 + lh * 16;
            kf0.q[0] = ((const uint4*)p0)[0]; kf0.q[1] = ((const uint4*)p0)[1];
            kf1.q[0] = ((const uint4*)p1)[0]; kf1.q[1] = ((const uint4*)p1)[1];
            s0 = wmma_bf16(qf[dc], kf0, s0);
            s1 = wmma_bf16(qf[dc], kf1, s1);
        }
        // streaming softmax per row
        for (int r = 0; r < 8; r++) {
            int rowg = q0 + r + (lh ? 8 : 0);
            int c0 = kb + lr, c1 = c0 + 16;
            float x0 = s0[r] * scale + (c0 > rowg ? -1e9f : 0.f);
            float x1 = s1[r] * scale + (c1 > rowg ? -1e9f : 0.f);
            float mx = fmaxf(x0, x1);
            for (int off = 1; off < 16; off <<= 1) mx = fmaxf(mx, __shfl_xor(mx, off, 32));
            float mn = fmaxf(m[r], mx);
            float sc = __expf(m[r] - mn);
            float p0 = __expf(x0 - mn);
            float p1 = __expf(x1 - mn);
            float rs = p0 + p1;
            for (int off = 1; off < 16; off <<= 1) rs += __shfl_xor(rs, off, 32);
            l[r] = l[r] * sc + rs;
            m[r] = mn;
            scarr[r] = sc;
            int rl = r + (lh ? 8 : 0);
            myP[rl * 32 + lr]      = f2bf(p0);
            myP[rl * 32 + 16 + lr] = f2bf(p1);
        }
        for (int nt = 0; nt < 8; nt++)
            for (int r = 0; r < 8; r++) O[nt][r] *= scarr[r];
        // P * V  (wave-private LDS re-layout; same-wave DS ops are in order)
        Frag pf;
        const unsigned short* pp = myP + lr * 32 + lh * 16;
        pf.q[0] = *(const uint4*)(pp);
        pf.q[1] = *(const uint4*)(pp + 8);
        for (int nt = 0; nt < 8; nt++) {
            Frag vf;
            const unsigned short* vb =
                v_t + ((size_t)bh * D_ + nt * 16 + lr) * S_ + kb + lh * 16;
            vf.q[0] = ((const uint4*)vb)[0];
            vf.q[1] = ((const uint4*)vb)[1];
            O[nt] = wmma_bf16(pf, vf, O[nt]);
        }
    }

    // epilogue: combined = g * (sig_q @ M)/den + (1-g) * O/l
    float g = 1.f / (1.f + __expf(-beta[0]));
    for (int nt = 0; nt < 8; nt++)
        for (int r = 0; r < 8; r++) O[nt][r] *= (1.f - g) / l[r];

    // pre-scale sig_q fragment by g/den (A-layout lane row is lr for all lanes)
    float sden = g / mem_den[(size_t)bh * S_ + q0 + lr];
    Frag sf[4];
    const unsigned short* sqb = sig_q + ((size_t)bh * S_ + q0 + lr) * D_;
    for (int dc = 0; dc < 4; dc++) {
        sf[dc].q[0] = *(const uint4*)(sqb + dc * 32 + lh * 16);
        sf[dc].q[1] = *(const uint4*)(sqb + dc * 32 + 8 + lh * 16);
        for (int e = 0; e < 16; e++)
            sf[dc].us[e] = f2bf(bf2f(sf[dc].us[e]) * sden);
    }
    for (int nt = 0; nt < 8; nt++) {
        for (int dc = 0; dc < 4; dc++) {
            Frag mf;
            const unsigned short* mb =
                M_t + ((size_t)h * D_ + nt * 16 + lr) * D_ + dc * 32 + lh * 16;
            mf.q[0] = ((const uint4*)mb)[0];
            mf.q[1] = ((const uint4*)mb)[1];
            O[nt] = wmma_bf16(sf[dc], mf, O[nt]);
        }
    }
    // store combined as bf16 in [B,S,H*D] for the Wo GEMM
    for (int nt = 0; nt < 8; nt++) {
        for (int r = 0; r < 8; r++) {
            int srow = q0 + r + (lh ? 8 : 0);
            comb[((size_t)b * S_ + srow) * HID_ + h * D_ + nt * 16 + lr] = f2bf(O[nt][r]);
        }
    }
}

// ---------------- M_new[h] = M[h] + sig_k^T @ v -----------------------------
// one wave per 16x16 tile; K loop over (b, s)
__global__ __launch_bounds__(32) void mem_update_kernel(
    const unsigned short* __restrict__ sig_k_t, const unsigned short* __restrict__ v_t,
    const float* __restrict__ M_in, float* __restrict__ M_out) {
    const int lane = threadIdx.x & 31;
    const int lr = lane & 15, lh = lane >> 4;
    const int d0 = blockIdx.x * 16, e0 = blockIdx.y * 16, h = blockIdx.z;

    v8f acc = vzero8();
    for (int r = 0; r < 8; r++) {
        int dd = d0 + r + (lh ? 8 : 0);
        acc[r] = M_in[((size_t)h * D_ + dd) * D_ + e0 + lr];
    }
    for (int b = 0; b < B_; b++) {
        size_t bh = (size_t)(b * H_ + h);
        for (int s0 = 0; s0 < S_; s0 += 32) {
            Frag af, bf;
            const unsigned short* pa = sig_k_t + (bh * D_ + d0 + lr) * S_ + s0 + lh * 16;
            af.q[0] = *(const uint4*)(pa);
            af.q[1] = *(const uint4*)(pa + 8);
            const unsigned short* pb = v_t + (bh * D_ + e0 + lr) * S_ + s0 + lh * 16;
            bf.q[0] = ((const uint4*)pb)[0];
            bf.q[1] = ((const uint4*)pb)[1];
            acc = wmma_bf16(af, bf, acc);
        }
    }
    for (int r = 0; r < 8; r++) {
        int dd = d0 + r + (lh ? 8 : 0);
        M_out[((size_t)h * D_ + dd) * D_ + e0 + lr] = acc[r];
    }
}

// ---------------- launch ----------------------------------------------------
extern "C" void kernel_launch(void* const* d_in, const int* in_sizes, int n_in,
                              void* d_out, int out_size, void* d_ws, size_t ws_size,
                              hipStream_t stream) {
    (void)in_sizes; (void)n_in; (void)out_size; (void)ws_size;
    const float* hs   = (const float*)d_in[0];
    const float* Wq   = (const float*)d_in[1];
    const float* bq   = (const float*)d_in[2];
    const float* Wk   = (const float*)d_in[3];
    const float* bk   = (const float*)d_in[4];
    const float* Wv   = (const float*)d_in[5];
    const float* bv   = (const float*)d_in[6];
    const float* Wo   = (const float*)d_in[7];
    const float* beta = (const float*)d_in[8];
    const float* M    = (const float*)d_in[9];
    const float* z    = (const float*)d_in[10];
    // d_in[11] attention_mask: pure causal -> applied analytically
    // d_in[12] position_ids: arange -> s used directly

    float* out   = (float*)d_out;
    float* fo    = out;                                  // [B,S,HID]
    float* M_out = out + (size_t)BS_ * HID_;             // [H,D,D]
    float* z_out = M_out + (size_t)H_ * D_ * D_;         // [H,D]

    char* ws = (char*)d_ws;
    size_t off = 0;
    auto carve = [&](size_t bytes) -> char* {
        char* p = ws + off;
        off += (bytes + 255) & ~(size_t)255;
        return p;
    };
    unsigned short* hs_b    = (unsigned short*)carve((size_t)BS_ * HID_ * 2);
    unsigned short* Wq_t    = (unsigned short*)carve((size_t)HID_ * HID_ * 2);
    unsigned short* Wk_t    = (unsigned short*)carve((size_t)HID_ * HID_ * 2);
    unsigned short* Wv_t    = (unsigned short*)carve((size_t)HID_ * HID_ * 2);
    unsigned short* Wo_t    = (unsigned short*)carve((size_t)HID_ * HID_ * 2);
    unsigned short* M_t     = (unsigned short*)carve((size_t)H_ * D_ * D_ * 2);
    float*          qp      = (float*)carve((size_t)BS_ * HID_ * 4);
    float*          kp      = (float*)carve((size_t)BS_ * HID_ * 4);
    float*          vp      = (float*)carve((size_t)BS_ * HID_ * 4);
    unsigned short* q_r     = (unsigned short*)carve((size_t)BHSD * 2);
    unsigned short* k_r     = (unsigned short*)carve((size_t)BHSD * 2);
    unsigned short* sig_q   = (unsigned short*)carve((size_t)BHSD * 2);
    unsigned short* sig_k_t = (unsigned short*)carve((size_t)BHSD * 2);
    unsigned short* v_t     = (unsigned short*)carve((size_t)BHSD * 2);
    float*          mden    = (float*)carve((size_t)B_ * H_ * S_ * 4);
    unsigned short* comb    = (unsigned short*)carve((size_t)BS_ * HID_ * 2);

    // 1) bf16 conversions (+ transposes so B-fragments are K-contiguous)
    int n1 = BS_ * HID_;
    cvt_bf16_kernel<<<n1 / 256, 256, 0, stream>>>(hs, hs_b, n1);
    int nw = HID_ * HID_;
    cvt_bf16_T_kernel<<<dim3(nw / 256, 1), 256, 0, stream>>>(Wq, Wq_t, HID_, HID_);
    cvt_bf16_T_kernel<<<dim3(nw / 256, 1), 256, 0, stream>>>(Wk, Wk_t, HID_, HID_);
    cvt_bf16_T_kernel<<<dim3(nw / 256, 1), 256, 0, stream>>>(Wv, Wv_t, HID_, HID_);
    cvt_bf16_T_kernel<<<dim3(nw / 256, 1), 256, 0, stream>>>(Wo, Wo_t, HID_, HID_);
    cvt_bf16_T_kernel<<<dim3((D_ * D_) / 256, H_), 256, 0, stream>>>(M, M_t, D_, D_);

    // 2) QKV projections (WMMA GEMM, 3-buffer async LDS pipeline)
    dim3 gg(HID_ / 128, BS_ / 128);
    gemm_bf16_kernel<<<gg, 256, 0, stream>>>(hs_b, Wq_t, bq, qp, BS_, HID_, HID_);
    gemm_bf16_kernel<<<gg, 256, 0, stream>>>(hs_b, Wk_t, bk, kp, BS_, HID_, HID_);
    gemm_bf16_kernel<<<gg, 256, 0, stream>>>(hs_b, Wv_t, bv, vp, BS_, HID_, HID_);

    // 3) RoPE + ELU features + layout shuffles + retrieval denominator
    rope_feat_kernel<<<B_ * H_ * S_, 128, 0, stream>>>(
        qp, kp, vp, z, q_r, k_r, sig_q, sig_k_t, v_t, mden);

    // 4) z_new
    znew_kernel<<<H_ * D_, 256, 0, stream>>>(sig_k_t, z, z_out);

    // 5) fused flash attention + memory retrieval -> combined (bf16)
    attn_ret_kernel<<<(B_ * H_ * (S_ / 16)) / 8, 256, 0, stream>>>(
        q_r, k_r, v_t, sig_q, M_t, mden, beta, comb);

    // 6) M_new (WMMA outer-product accumulation)
    mem_update_kernel<<<dim3(D_ / 16, D_ / 16, H_), 32, 0, stream>>>(
        sig_k_t, v_t, M, M_out);

    // 7) output projection
    gemm_bf16_kernel<<<gg, 256, 0, stream>>>(comb, Wo_t, nullptr, fo, BS_, HID_, HID_);
}
